// F2MF_multihead_27882927685679
// MI455X (gfx1250) — matrix operand order repaired and checked by reading, not compiled, not run on hardware
//
#include <hip/hip_runtime.h>
#include <hip/hip_bf16.h>
#include <math.h>

#define HH 64
#define WW 64
#define NPIX 4096

typedef __attribute__((ext_vector_type(16))) __bf16 v16bf;
typedef __attribute__((ext_vector_type(8)))  float  v8f;

// ---------------- device helpers ----------------

__device__ __forceinline__ unsigned short f2bf(float f) {
  unsigned int u = __float_as_uint(f);
  unsigned int r = (u + 0x7fffu + ((u >> 16) & 1u)) >> 16;  // round-to-nearest-even
  return (unsigned short)r;
}

__device__ __forceinline__ float bilin(const float* __restrict__ ch, float py, float px) {
  float y0f = floorf(py), x0f = floorf(px);
  int y0 = (int)y0f, x0 = (int)x0f;
  float wy = py - y0f, wx = px - x0f;
  float acc = 0.f;
#pragma unroll
  for (int dy = 0; dy < 2; ++dy) {
#pragma unroll
    for (int dx = 0; dx < 2; ++dx) {
      int yy = y0 + dy, xx = x0 + dx;
      if (yy >= 0 && yy < HH && xx >= 0 && xx < WW) {
        float w = (dy ? wy : 1.f - wy) * (dx ? wx : 1.f - wx);
        acc += w * ch[(yy << 6) + xx];
      }
    }
  }
  return acc;
}

// ---------------- elementwise kernels ----------------

__global__ void bn_relu_k(const float* __restrict__ x1, int C1, const float* __restrict__ x2,
                          const float* __restrict__ g, const float* __restrict__ b,
                          const float* __restrict__ m, const float* __restrict__ v,
                          float* __restrict__ h, int total) {
  int idx = blockIdx.x * blockDim.x + threadIdx.x;
  if (idx >= total) return;
  int c = idx >> 12;
  float xv = (c < C1) ? x1[idx] : x2[idx - (size_t)C1 * NPIX];
  float s = g[c] * (xv - m[c]) * rsqrtf(v[c] + 1e-5f) + b[c];
  h[idx] = s > 0.f ? s : 0.f;
}

// Pack im2col patches straight into WMMA B-fragment layout:
// PB[((pt*KB+kb)*32+lane)*16+j] = P[k = kb*32 + (lane>>4)*16 + j][pix = pt*16 + (lane&15)]
__global__ void pack_im2col_k(const float* __restrict__ h, unsigned short* __restrict__ PB,
                              int K, int Kdim, int KB, int total) {
  int idx = blockIdx.x * blockDim.x + threadIdx.x;
  if (idx >= total) return;
  int j = idx & 15, l = (idx >> 4) & 31;
  int t = idx >> 9;
  int kb = t % KB, pt = t / KB;
  int kdim = kb * 32 + ((l >> 4) << 4) + j;
  int pix = pt * 16 + (l & 15);
  float val = 0.f;
  if (kdim < Kdim) {
    int KK = K * K;
    int c = kdim / KK, k = kdim % KK;
    int pad = (K - 1) >> 1;
    int y = pix >> 6, x = pix & 63;
    int sy = y + k / K - pad, sx = x + k % K - pad;
    if (sy >= 0 && sy < HH && sx >= 0 && sx < WW)
      val = h[(size_t)c * NPIX + (sy << 6) + sx];
  }
  PB[idx] = f2bf(val);
}

// Deformable sampling into the same B-fragment layout. off: [2*KK][NPIX], (dy,dx) per k.
__global__ void pack_deform_k(const float* __restrict__ h, const float* __restrict__ off,
                              unsigned short* __restrict__ PB,
                              int K, int Kdim, int KB, int total) {
  int idx = blockIdx.x * blockDim.x + threadIdx.x;
  if (idx >= total) return;
  int j = idx & 15, l = (idx >> 4) & 31;
  int t = idx >> 9;
  int kb = t % KB, pt = t / KB;
  int kdim = kb * 32 + ((l >> 4) << 4) + j;
  int pix = pt * 16 + (l & 15);
  float val = 0.f;
  if (kdim < Kdim) {
    int KK = K * K;
    int c = kdim / KK, k = kdim % KK;
    int pad = (K - 1) >> 1;
    int y = pix >> 6, x = pix & 63;
    float py = (float)(y + k / K - pad) + off[(size_t)(2 * k) * NPIX + pix];
    float px = (float)(x + k % K - pad) + off[(size_t)(2 * k + 1) * NPIX + pix];
    val = bilin(h + (size_t)c * NPIX, py, px);
  }
  PB[idx] = f2bf(val);
}

// Pack weights [OC][Kdim] into WMMA A-fragment layout (16x32 bf16 per tile):
// lane<16 holds row m, K {0..15}; lane>=16 holds K {16..31}
// Tile count is padded (zeros) so the GEMM can always read oc-tile pairs.
__global__ void pack_w_k(const float* __restrict__ w, unsigned short* __restrict__ WA,
                         int OC, int Kdim, int KB, int total) {
  int idx = blockIdx.x * blockDim.x + threadIdx.x;
  if (idx >= total) return;
  int j = idx & 15, l = (idx >> 4) & 31;
  int t = idx >> 9;
  int kb = t % KB, ocb = t / KB;
  int oc = ocb * 16 + (l & 15);
  int kdim = kb * 32 + ((l >> 4) << 4) + j;
  float val = (oc < OC && kdim < Kdim) ? w[(size_t)oc * Kdim + kdim] : 0.f;
  WA[idx] = f2bf(val);
}

// ---------------- WMMA GEMM: out[oc][pix] = sum_k W[oc][k] * P[k][pix] (+bias) -------------
// One wave per (32 oc) x (64 pixel) strip: 2 A-frags + 4 B-frags -> 8 WMMAs per k-step of 32.
__global__ __launch_bounds__(32) void gemm_k(const unsigned short* __restrict__ WAu,
                                             const unsigned short* __restrict__ PBu,
                                             float* __restrict__ out,
                                             const float* __restrict__ bias,
                                             int OC, int KB) {
  const int l = threadIdx.x;
  const int ocp = blockIdx.y;           // oc tile pair index
  const int pt0 = blockIdx.x * 4;
  const v16bf* __restrict__ A0 = (const v16bf*)WAu + ((size_t)(2 * ocp + 0) * KB) * 32 + l;
  const v16bf* __restrict__ A1 = (const v16bf*)WAu + ((size_t)(2 * ocp + 1) * KB) * 32 + l;
  const v16bf* __restrict__ B0 = (const v16bf*)PBu + ((size_t)(pt0 + 0) * KB) * 32 + l;
  const v16bf* __restrict__ B1 = (const v16bf*)PBu + ((size_t)(pt0 + 1) * KB) * 32 + l;
  const v16bf* __restrict__ B2 = (const v16bf*)PBu + ((size_t)(pt0 + 2) * KB) * 32 + l;
  const v16bf* __restrict__ B3 = (const v16bf*)PBu + ((size_t)(pt0 + 3) * KB) * 32 + l;
  v8f c00 = {}, c01 = {}, c02 = {}, c03 = {};
  v8f c10 = {}, c11 = {}, c12 = {}, c13 = {};
  for (int kb = 0; kb < KB; ++kb) {
    size_t o = (size_t)kb * 32;
    v16bf a0 = A0[o], a1 = A1[o];
    v16bf b0 = B0[o], b1 = B1[o], b2 = B2[o], b3 = B3[o];
    if (kb + 1 < KB) {
      __builtin_prefetch((const void*)(B0 + o + 32), 0, 1);
      __builtin_prefetch((const void*)(B1 + o + 32), 0, 1);
      __builtin_prefetch((const void*)(B2 + o + 32), 0, 1);
      __builtin_prefetch((const void*)(B3 + o + 32), 0, 1);
    }
    c00 = __builtin_amdgcn_wmma_f32_16x16x32_bf16(false, a0, false, b0, (short)0, c00, false, false);
    c01 = __builtin_amdgcn_wmma_f32_16x16x32_bf16(false, a0, false, b1, (short)0, c01, false, false);
    c02 = __builtin_amdgcn_wmma_f32_16x16x32_bf16(false, a0, false, b2, (short)0, c02, false, false);
    c03 = __builtin_amdgcn_wmma_f32_16x16x32_bf16(false, a0, false, b3, (short)0, c03, false, false);
    c10 = __builtin_amdgcn_wmma_f32_16x16x32_bf16(false, a1, false, b0, (short)0, c10, false, false);
    c11 = __builtin_amdgcn_wmma_f32_16x16x32_bf16(false, a1, false, b1, (short)0, c11, false, false);
    c12 = __builtin_amdgcn_wmma_f32_16x16x32_bf16(false, a1, false, b2, (short)0, c12, false, false);
    c13 = __builtin_amdgcn_wmma_f32_16x16x32_bf16(false, a1, false, b3, (short)0, c13, false, false);
  }
  const int n = l & 15;
  const int mh = (l >> 4) * 8;  // lanes 0-15: rows 0..7; lanes 16-31: rows 8..15
#pragma unroll
  for (int r = 0; r < 8; ++r) {
    int oc0 = ocp * 32 + mh + r;
    int oc1 = oc0 + 16;
    if (oc0 < OC) {
      float bb = bias ? bias[oc0] : 0.f;
      out[(size_t)oc0 * NPIX + (pt0 + 0) * 16 + n] = c00[r] + bb;
      out[(size_t)oc0 * NPIX + (pt0 + 1) * 16 + n] = c01[r] + bb;
      out[(size_t)oc0 * NPIX + (pt0 + 2) * 16 + n] = c02[r] + bb;
      out[(size_t)oc0 * NPIX + (pt0 + 3) * 16 + n] = c03[r] + bb;
    }
    if (oc1 < OC) {
      float bb = bias ? bias[oc1] : 0.f;
      out[(size_t)oc1 * NPIX + (pt0 + 0) * 16 + n] = c10[r] + bb;
      out[(size_t)oc1 * NPIX + (pt0 + 1) * 16 + n] = c11[r] + bb;
      out[(size_t)oc1 * NPIX + (pt0 + 2) * 16 + n] = c12[r] + bb;
      out[(size_t)oc1 * NPIX + (pt0 + 3) * 16 + n] = c13[r] + bb;
    }
  }
}

// ---------------- correlation / tail kernels ----------------

__global__ void l2norm_k(const float* __restrict__ g, float* __restrict__ o) {
  int pix = blockIdx.x * blockDim.x + threadIdx.x;
  if (pix >= NPIX) return;
  float s = 0.f;
  for (int c = 0; c < 128; ++c) { float v = g[(size_t)c * NPIX + pix]; s += v * v; }
  float inv = rsqrtf(s);
  for (int c = 0; c < 128; ++c) o[(size_t)c * NPIX + pix] = g[(size_t)c * NPIX + pix] * inv;
}

__global__ void corr_k(const float* __restrict__ f1, const float* __restrict__ f2,
                       float* __restrict__ xout, int base) {
  int idx = blockIdx.x * blockDim.x + threadIdx.x;
  if (idx >= 81 * NPIX) return;
  int pix = idx & (NPIX - 1);
  int pk = idx >> 12;
  int dy = pk / 9, dx = pk % 9;
  int y = pix >> 6, x = pix & 63;
  int sy = y + dy * 2 - 8, sx = x + dx * 2 - 8;  // dilation_patch=2, md=8
  float s = 0.f;
  if (sy >= 0 && sy < HH && sx >= 0 && sx < WW) {
    int sp = (sy << 6) + sx;
    for (int c = 0; c < 128; ++c)
      s += f1[(size_t)c * NPIX + pix] * f2[(size_t)c * NPIX + sp];
  }
  xout[(size_t)(base + pk) * NPIX + pix] = s;
}

__global__ void softmax5_k(const float* __restrict__ w, float* __restrict__ pr) {
  int pix = blockIdx.x * blockDim.x + threadIdx.x;
  if (pix >= NPIX) return;
  float mx = -1e30f;
  for (int j = 0; j < 5; ++j) mx = fmaxf(mx, w[(size_t)j * NPIX + pix]);
  float e[5], s = 0.f;
  for (int j = 0; j < 5; ++j) { e[j] = expf(w[(size_t)j * NPIX + pix] - mx); s += e[j]; }
  float inv = 1.f / s;
  for (int j = 0; j < 5; ++j) pr[(size_t)j * NPIX + pix] = e[j] * inv;
}

__global__ void combine_k(const float* __restrict__ sem, const float* __restrict__ rgb,
                          const float* __restrict__ flow, const float* __restrict__ probs,
                          const float* __restrict__ psem, const float* __restrict__ prgb,
                          float* __restrict__ out) {
  int idx = blockIdx.x * blockDim.x + threadIdx.x;
  if (idx >= 384 * NPIX) return;
  int pix = idx & (NPIX - 1);
  int c = idx >> 12;
  int y = pix >> 6, x = pix & 63;
  int cc; const float* frames; const float* pred; int CF;
  if (c < 128) { cc = c;       frames = sem; pred = psem; CF = 128; }
  else         { cc = c - 128; frames = rgb; pred = prgb; CF = 256; }
  float acc = probs[pix] * pred[(size_t)cc * NPIX + pix];
#pragma unroll
  for (int f = 0; f < 4; ++f) {
    float fx = flow[(size_t)(2 * f) * NPIX + pix];
    float fy = flow[(size_t)(2 * f + 1) * NPIX + pix];
    const float* ch = frames + (size_t)(f * CF + cc) * NPIX;
    acc += probs[(size_t)(f + 1) * NPIX + pix] * bilin(ch, (float)y + fy, (float)x + fx);
  }
  out[(size_t)c * NPIX + pix] = acc;  // sem rows 0..127, rgb rows 128..383 (flat concat)
}

// ---------------- host side ----------------

struct BNP { const float *g, *b, *m, *v; };
struct BlkP { BNP bn; const float *ow, *ob, *w, *b; };

static void run_dconv(hipStream_t s, const BlkP& bk, const float* x1, int C1, const float* x2,
                      int C, int K, int OC, float* h, unsigned short* PB, unsigned short* WA,
                      float* off, float* outp) {
  int KK = K * K, Kdim = C * KK, KB = (Kdim + 31) / 32;
  int tot = C * NPIX;
  bn_relu_k<<<(tot + 255) / 256, 256, 0, s>>>(x1, C1, x2, bk.bn.g, bk.bn.b, bk.bn.m, bk.bn.v, h, tot);
  tot = KB * 131072;
  pack_im2col_k<<<(tot + 255) / 256, 256, 0, s>>>(h, PB, K, Kdim, KB, tot);
  // offset conv (OC = 2*K*K), pad tiles even for pair-GEMM
  int OCo = 2 * KK, ot = ((OCo + 15) / 16 + 1) & ~1;
  int wt = ot * KB * 512;
  pack_w_k<<<(wt + 255) / 256, 256, 0, s>>>(bk.ow, WA, OCo, Kdim, KB, wt);
  { dim3 g(64, ot / 2); gemm_k<<<g, 32, 0, s>>>(WA, PB, off, bk.ob, OCo, KB); }
  tot = KB * 131072;
  pack_deform_k<<<(tot + 255) / 256, 256, 0, s>>>(h, off, PB, K, Kdim, KB, tot);
  int mt = ((OC + 15) / 16 + 1) & ~1;
  wt = mt * KB * 512;
  pack_w_k<<<(wt + 255) / 256, 256, 0, s>>>(bk.w, WA, OC, Kdim, KB, wt);
  { dim3 g(64, mt / 2); gemm_k<<<g, 32, 0, s>>>(WA, PB, outp, bk.b, OC, KB); }
}

extern "C" void kernel_launch(void* const* d_in, const int* in_sizes, int n_in,
                              void* d_out, int out_size, void* d_ws, size_t ws_size,
                              hipStream_t stream) {
  (void)n_in; (void)out_size; (void)ws_size;
  // ---- input detection (insertion vs pytree-sorted flattening) ----
  bool sem_first = (in_sizes[0] == 512 * NPIX);
  const float* in_sem = (const float*)d_in[sem_first ? 0 : 1];
  const float* in_rgb = (const float*)d_in[sem_first ? 1 : 0];
  bool sorted = (in_sizes[2] == 128);  // sorted: corr_bn.beta(128) first; insertion: fusion.bn.gamma(1536)
  int cur = 2;
  auto nxt = [&]() { return (const float*)d_in[cur++]; };
  BNP corr_bn{}; const float* corr_w = nullptr;
  BlkP fusion{}, f2m{}, fsem{}, frgb{}, wgt{}, shared[6]{};
  auto rd_bn = [&](BNP& bn) {
    if (sorted) { bn.b = nxt(); bn.g = nxt(); bn.m = nxt(); bn.v = nxt(); }
    else        { bn.g = nxt(); bn.b = nxt(); bn.m = nxt(); bn.v = nxt(); }
  };
  auto rd_blk = [&](BlkP& k) {
    if (sorted) { k.b = nxt(); rd_bn(k.bn); k.ob = nxt(); k.ow = nxt(); k.w = nxt(); }
    else        { rd_bn(k.bn); k.ow = nxt(); k.ob = nxt(); k.w = nxt(); k.b = nxt(); }
  };
  if (sorted) {
    rd_bn(corr_bn); corr_w = nxt();
    rd_blk(frgb); rd_blk(fsem); rd_blk(f2m); rd_blk(fusion);
    for (int i = 0; i < 6; ++i) rd_blk(shared[i]);
    rd_blk(wgt);
  } else {
    rd_blk(fusion); rd_bn(corr_bn); corr_w = nxt();
    for (int i = 0; i < 6; ++i) rd_blk(shared[i]);
    rd_blk(f2m); rd_blk(fsem); rd_blk(frgb); rd_blk(wgt);
  }
  // ---- workspace bump allocation ----
  char* p = (char*)d_ws;
  auto alloc = [&](size_t bytes) -> void* {
    void* r = (void*)p; p += (bytes + 255) & ~(size_t)255; return r;
  };
  float*          h     = (float*)alloc((size_t)1536 * NPIX * 4);
  float*          off   = (float*)alloc((size_t)18 * NPIX * 4);
  unsigned short* PB    = (unsigned short*)alloc((size_t)5664 * NPIX * 2);
  unsigned short* WA    = (unsigned short*)alloc((size_t)512 * 5664 * 2);
  float*          gout  = (float*)alloc((size_t)128 * NPIX * 4);
  float*          xb    = (float*)alloc((size_t)627 * NPIX * 4);
  float*          xb2   = (float*)alloc((size_t)512 * NPIX * 4);
  float*          convf = (float*)alloc((size_t)4 * 128 * NPIX * 4);
  float*          flow  = (float*)alloc((size_t)8 * NPIX * 4);
  float*          psem  = (float*)alloc((size_t)128 * NPIX * 4);
  float*          prgb  = (float*)alloc((size_t)256 * NPIX * 4);
  float*          pw5   = (float*)alloc((size_t)5 * NPIX * 4);
  float*          probs = (float*)alloc((size_t)5 * NPIX * 4);

  // ---- fusion block: concat(sem,rgb) 1536 -> 384, K=1 ----
  run_dconv(stream, fusion, in_sem, 512, in_rgb, 1536, 1, 384, h, PB, WA, off, xb);

  // ---- correlation branch ----
  {
    int Kdim = 1152, KB = 36;
    int wt = 8 * KB * 512;  // 8 oc tiles (even)
    pack_w_k<<<(wt + 255) / 256, 256, 0, stream>>>(corr_w, WA, 128, Kdim, KB, wt);
    for (int i = 0; i < 4; ++i) {
      const float* f = in_sem + (size_t)i * 128 * NPIX;
      int tot = 128 * NPIX;
      bn_relu_k<<<(tot + 255) / 256, 256, 0, stream>>>(f, 128, f, corr_bn.g, corr_bn.b,
                                                       corr_bn.m, corr_bn.v, h, tot);
      tot = KB * 131072;
      pack_im2col_k<<<(tot + 255) / 256, 256, 0, stream>>>(h, PB, 3, Kdim, KB, tot);
      dim3 g(64, 4);
      gemm_k<<<g, 32, 0, stream>>>(WA, PB, gout, nullptr, 128, KB);
      l2norm_k<<<16, 256, 0, stream>>>(gout, convf + (size_t)i * 128 * NPIX);
    }
    for (int i = 0; i < 3; ++i) {
      int tot = 81 * NPIX;
      corr_k<<<(tot + 255) / 256, 256, 0, stream>>>(convf + (size_t)i * 128 * NPIX,
                                                    convf + (size_t)(i + 1) * 128 * NPIX,
                                                    xb, 384 + i * 81);
    }
  }

  // ---- 6 shared dconv blocks (627->512, then 512->512), ping-pong xb/xb2 ----
  float* a = xb; float* bout = xb2;
  int C = 627;
  for (int bi = 0; bi < 6; ++bi) {
    run_dconv(stream, shared[bi], a, C, a, C, 3, 512, h, PB, WA, off, bout);
    float* t = a; a = bout; bout = t;
    C = 512;
  }
  // final features in `a` (512 channels)

  // ---- heads ----
  run_dconv(stream, f2m,  a, 512, a, 512, 3, 8,   h, PB, WA, off, flow);
  run_dconv(stream, fsem, a, 512, a, 512, 3, 128, h, PB, WA, off, psem);
  run_dconv(stream, frgb, a, 512, a, 512, 3, 256, h, PB, WA, off, prgb);
  run_dconv(stream, wgt,  a, 512, a, 512, 3, 5,   h, PB, WA, off, pw5);

  // ---- softmax + warp + combine ----
  softmax5_k<<<16, 256, 0, stream>>>(pw5, probs);
  int tot = 384 * NPIX;
  combine_k<<<(tot + 255) / 256, 256, 0, stream>>>(in_sem, in_rgb, flow, probs, psem, prgb,
                                                   (float*)d_out);
}
